// QuantizadorVetorial_63763084476448
// MI455X (gfx1250) — matrix operand order repaired and checked by reading, not compiled, not run on hardware
//
#include <hip/hip_runtime.h>
#include <stdint.h>

// Problem sizes
static constexpr int D    = 512;      // feature dim
static constexpr int NROW = 32768;    // 32*1024 rows
static constexpr int K    = 8192;     // codebook size
static constexpr int MT   = 128;      // rows per workgroup tile
static constexpr int NT   = 128;      // codes per k-tile
static constexpr int DC   = 32;       // depth per bf16 WMMA chunk
static constexpr int LSTR = 40;       // LDS row stride in shorts (80B = 64B tile row + 16B TDM pad)
static constexpr int PLANE_B = MT * LSTR * 2;      // 10240 B per tile plane
static constexpr int TOTAL_STEPS = (K / NT) * (D / DC);  // 64 * 16 = 1024

typedef __attribute__((ext_vector_type(16))) __bf16 v16bf;
typedef __attribute__((ext_vector_type(8)))  float  v8f;
typedef __attribute__((ext_vector_type(4)))  unsigned int tdm_v4u;
typedef __attribute__((ext_vector_type(8)))  int          tdm_v8i;
typedef __attribute__((ext_vector_type(4)))  int          tdm_v4i;

#if defined(__has_builtin)
#if __has_builtin(__builtin_amdgcn_tensor_load_to_lds)
#define USE_TDM 1
#endif
#endif
#ifndef USE_TDM
#define USE_TDM 0
#endif

union Frag { v16bf v; uint4 u[2]; };

__device__ __forceinline__ uint16_t f2bf(float f) {
    uint32_t u = __float_as_uint(f);
    u += 0x7FFFu + ((u >> 16) & 1u);          // round-to-nearest-even
    return (uint16_t)(u >> 16);
}
__device__ __forceinline__ float bf2f(uint16_t h) {
    return __uint_as_float(((uint32_t)h) << 16);
}

__device__ __forceinline__ void wait_tensorcnt0() {
#if defined(__has_builtin)
#if __has_builtin(__builtin_amdgcn_s_wait_tensorcnt)
    __builtin_amdgcn_s_wait_tensorcnt(0);
    return;
#else
    asm volatile("s_wait_tensorcnt 0x0" ::: "memory");
    return;
#endif
#else
    asm volatile("s_wait_tensorcnt 0x0" ::: "memory");
#endif
}

#if USE_TDM
// TDM: DMA one 2D tile (128 rows x 32 bf16) from a row-major [nrows][512] bf16
// plane into LDS, inserting 16B padding after each 64B row (LDS stride 80B).
__device__ __forceinline__ void tdm_tile_load(uint32_t lds_addr, const uint16_t* gsrc,
                                              uint32_t nrows_tensor) {
    uint64_t ga = (uint64_t)(uintptr_t)gsrc;
    // group0: count=1 | lds_addr | global_addr[56:0] | type=2
    tdm_v4u g0 = { 1u,
                   lds_addr,
                   (uint32_t)ga,
                   (uint32_t)((ga >> 32) & 0x01FFFFFFu) | (2u << 30) };
    // group1: data_size=2B, pad_enable, pad_interval=16DW(64B), pad_amount=4DW(16B)
    //         tensor_dim0=512, tensor_dim1=nrows, tile_dim0=32, tile_dim1=128,
    //         tensor_dim0_stride=512
    tdm_v8i g1 = {
        (int)((1u << 16) | (1u << 20) | (3u << 22) | (3u << 25)),
        (int)(512u << 16),                                   // tensor_dim0 lo16 @ [63:48]
        (int)((nrows_tensor & 0xFFFFu) << 16),               // dim0 hi=0 | tensor_dim1 lo16
        (int)(((nrows_tensor >> 16) & 0xFFFFu) | (32u << 16)), // tensor_dim1 hi | tile_dim0
        (int)(128u),                                         // tile_dim1=128 | tile_dim2=0
        (int)(512),                                          // tensor_dim0_stride lo32
        0, 0 };                                              // stride hi / dim1_stride = 0
    tdm_v4i gz4a = { 0, 0, 0, 0 };                           // group2 (2D: unused)
    tdm_v4i gz4b = { 0, 0, 0, 0 };                           // group3 (2D: unused)
    tdm_v8i gz8  = { 0, 0, 0, 0, 0, 0, 0, 0 };
    __builtin_amdgcn_tensor_load_to_lds(g0, g1, gz4a, gz4b, gz8, 0);
}
#endif

// ---- Kernel A: split x into bf16 hi/lo planes -------------------------------
__global__ void vq_cvt_x(const float* __restrict__ x,
                         uint16_t* __restrict__ xhi, uint16_t* __restrict__ xlo) {
    int t = blockIdx.x * 256 + threadIdx.x;   // one float4 per thread
    float4 v = ((const float4*)x)[t];
    float f[4] = {v.x, v.y, v.z, v.w};
    uint32_t hh[4], ll[4];
#pragma unroll
    for (int i = 0; i < 4; ++i) {
        uint16_t h = f2bf(f[i]);
        uint16_t l = f2bf(f[i] - bf2f(h));
        hh[i] = h; ll[i] = l;
    }
    uint2 ph = { hh[0] | (hh[1] << 16), hh[2] | (hh[3] << 16) };
    uint2 pl = { ll[0] | (ll[1] << 16), ll[2] | (ll[3] << 16) };
    ((uint2*)xhi)[t] = ph;
    ((uint2*)xlo)[t] = pl;
}

// ---- Kernel B: split weight into bf16 hi/lo planes + ||w_k||^2 (f32) --------
__global__ void vq_cvt_w(const float* __restrict__ w,
                         uint16_t* __restrict__ whi, uint16_t* __restrict__ wlo,
                         float* __restrict__ wnorm) {
    int lane = threadIdx.x & 31;
    int wave = threadIdx.x >> 5;
    int code = blockIdx.x * 8 + wave;         // one wave per code row
    const float* row = w + (size_t)code * D;
    float s = 0.f;
#pragma unroll
    for (int j = 0; j < 16; ++j) {
        int d = j * 32 + lane;                // coalesced
        float f = row[d];
        s += f * f;
        uint16_t h = f2bf(f);
        uint16_t l = f2bf(f - bf2f(h));
        whi[(size_t)code * D + d] = h;
        wlo[(size_t)code * D + d] = l;
    }
#pragma unroll
    for (int off = 16; off; off >>= 1) s += __shfl_xor(s, off, 32);
    if (lane == 0) wnorm[code] = s;
}

// ---- Kernel C: zero the loss accumulator ------------------------------------
__global__ void vq_zero(float* loss) {
    if (threadIdx.x == 0) loss[0] = 0.f;
}

// ---- Kernel D: fused GEMM(bf16x3 WMMA) + argmin + gather + loss -------------
__global__ __launch_bounds__(256)
void vq_main(const float* __restrict__ x, const float* __restrict__ weight,
             const uint16_t* __restrict__ xhi, const uint16_t* __restrict__ xlo,
             const uint16_t* __restrict__ whi, const uint16_t* __restrict__ wlo,
             const float* __restrict__ wnorm,
             float* __restrict__ out, float* __restrict__ loss) {
    // [buf][plane: 0=x_hi 1=x_lo 2=w_hi 3=w_lo][row*LSTR]
    __shared__ __align__(16) uint16_t sTiles[2][4][MT * LSTR];
    __shared__ float redv[MT][2];
    __shared__ int   redi[MT][2];
    __shared__ int   fidx[MT];
    __shared__ float lred[8];

    const int t    = threadIdx.x;
    const int lane = t & 31;
    const int wave = t >> 5;
    const int wm   = wave & 3;                // row group:   rows 32*wm .. +31
    const int wc   = wave >> 2;               // code half:   codes 64*wc .. +63
    const int row0g = blockIdx.x * MT;
    const int hl   = (lane < 16) ? 0 : 1;     // lane half selects K sub-range
    const int l15  = lane & 15;

    v8f   acc[2][4];
    float bestv[2][8];
    int   besti[2][8];
#pragma unroll
    for (int rf = 0; rf < 2; ++rf)
#pragma unroll
        for (int j = 0; j < 8; ++j) { bestv[rf][j] = 3.4e38f; besti[rf][j] = 0; }

#if USE_TDM
    const uint32_t lds_base = (uint32_t)(uintptr_t)(&sTiles[0][0][0]);
    // waves 0..3 each own one plane; issue the 2D tile DMA for step `stepN`
    auto tdm_issue = [&](int buf, int stepN) {
        int ktN = stepN >> 4;
        int d0N = (stepN & 15) * DC;
        uint32_t ldsa = lds_base + (uint32_t)((buf * 4 + wave) * PLANE_B);
        if (wave == 0)      tdm_tile_load(ldsa, xhi + (size_t)row0g * D + d0N, (uint32_t)NROW);
        else if (wave == 1) tdm_tile_load(ldsa, xlo + (size_t)row0g * D + d0N, (uint32_t)NROW);
        else if (wave == 2) tdm_tile_load(ldsa, whi + (size_t)ktN * NT * D + d0N, (uint32_t)K);
        else                tdm_tile_load(ldsa, wlo + (size_t)ktN * NT * D + d0N, (uint32_t)K);
    };
    if (wave < 4) tdm_issue(0, 0);            // prologue: prefetch step 0 into buf 0
#else
    const int srow = t >> 1;                  // manual staging coords
    const int scol = (t & 1) * 16;
#endif

    int step = 0;
    for (int kt = 0; kt < K / NT; ++kt) {
#pragma unroll
        for (int rf = 0; rf < 2; ++rf)
#pragma unroll
            for (int cf = 0; cf < 4; ++cf)
#pragma unroll
                for (int e = 0; e < 8; ++e) acc[rf][cf][e] = 0.f;

        for (int dsi = 0; dsi < D / DC; ++dsi) {
#if USE_TDM
            const int buf = step & 1;
            if (wave < 4) wait_tensorcnt0();  // own plane's DMA for `buf` done
            __syncthreads();                  // all planes ready; buf^1 fully consumed
            if (wave < 4 && step + 1 < TOTAL_STEPS)
                tdm_issue(buf ^ 1, step + 1); // prefetch next chunk under compute
#else
            const int buf = 0;
            const int d0 = dsi * DC;
            __syncthreads();
            {
                const uint4* shp = (const uint4*)(xhi + (size_t)(row0g + srow) * D + d0 + scol);
                const uint4* slp = (const uint4*)(xlo + (size_t)(row0g + srow) * D + d0 + scol);
                uint4* dh = (uint4*)(&sTiles[0][0][srow * LSTR + scol]);
                uint4* dl = (uint4*)(&sTiles[0][1][srow * LSTR + scol]);
                dh[0] = shp[0]; dh[1] = shp[1];
                dl[0] = slp[0]; dl[1] = slp[1];
                int cg = kt * NT + srow;
                const uint4* swh = (const uint4*)(whi + (size_t)cg * D + d0 + scol);
                const uint4* swl = (const uint4*)(wlo + (size_t)cg * D + d0 + scol);
                uint4* ch = (uint4*)(&sTiles[0][2][srow * LSTR + scol]);
                uint4* cl = (uint4*)(&sTiles[0][3][srow * LSTR + scol]);
                ch[0] = swh[0]; ch[1] = swh[1];
                cl[0] = swl[0]; cl[1] = swl[1];
            }
            __syncthreads();
#endif
            const uint16_t* sXh = &sTiles[buf][0][0];
            const uint16_t* sXl = &sTiles[buf][1][0];
            const uint16_t* sWh = &sTiles[buf][2][0];
            const uint16_t* sWl = &sTiles[buf][3][0];

            // A fragments: 16x32 bf16, lane<16: K 0..7 & 16..23; lane>=16: K 8..15 & 24..31
            Frag axh[2], axl[2];
#pragma unroll
            for (int rf = 0; rf < 2; ++rf) {
                int rowL = 32 * wm + 16 * rf + l15;
                axh[rf].u[0] = *(const uint4*)(sXh + rowL * LSTR + hl * 8);
                axh[rf].u[1] = *(const uint4*)(sXh + rowL * LSTR + 16 + hl * 8);
                axl[rf].u[0] = *(const uint4*)(sXl + rowL * LSTR + hl * 8);
                axl[rf].u[1] = *(const uint4*)(sXl + rowL * LSTR + 16 + hl * 8);
            }
            // B fragments: 32x16 bf16, column N=l15, lane<16: K 0..7 & 16..23; lane>=16: K 8..15 & 24..31
            Frag bwh[4], bwl[4];
#pragma unroll
            for (int cf = 0; cf < 4; ++cf) {
                int codeL = 64 * wc + 16 * cf + l15;
                bwh[cf].u[0] = *(const uint4*)(sWh + codeL * LSTR + hl * 8);
                bwh[cf].u[1] = *(const uint4*)(sWh + codeL * LSTR + 16 + hl * 8);
                bwl[cf].u[0] = *(const uint4*)(sWl + codeL * LSTR + hl * 8);
                bwl[cf].u[1] = *(const uint4*)(sWl + codeL * LSTR + 16 + hl * 8);
            }
            // bf16x3: dot ~= hi*hi + hi*lo + lo*hi   (f32 accumulate)
#pragma unroll
            for (int rf = 0; rf < 2; ++rf)
#pragma unroll
                for (int cf = 0; cf < 4; ++cf) {
                    acc[rf][cf] = __builtin_amdgcn_wmma_f32_16x16x32_bf16(
                        false, axh[rf].v, false, bwh[cf].v, (short)0, acc[rf][cf], false, false);
                    acc[rf][cf] = __builtin_amdgcn_wmma_f32_16x16x32_bf16(
                        false, axh[rf].v, false, bwl[cf].v, (short)0, acc[rf][cf], false, false);
                    acc[rf][cf] = __builtin_amdgcn_wmma_f32_16x16x32_bf16(
                        false, axl[rf].v, false, bwh[cf].v, (short)0, acc[rf][cf], false, false);
                }
            ++step;
        }

        // score = ||w||^2 - 2*dot ; update running per-row argmin
#pragma unroll
        for (int cf = 0; cf < 4; ++cf) {
            int code = kt * NT + 64 * wc + 16 * cf + l15;
            float wnv = wnorm[code];
#pragma unroll
            for (int rf = 0; rf < 2; ++rf)
#pragma unroll
                for (int j = 0; j < 8; ++j) {
                    float s = wnv - 2.f * acc[rf][cf][j];
                    if (s < bestv[rf][j]) { bestv[rf][j] = s; besti[rf][j] = code; }
                }
        }
    }

    // reduce (min,idx) across the 16 lanes that share each row
#pragma unroll
    for (int rf = 0; rf < 2; ++rf)
#pragma unroll
        for (int j = 0; j < 8; ++j) {
            float v = bestv[rf][j];
            int   ii = besti[rf][j];
#pragma unroll
            for (int off = 8; off; off >>= 1) {
                float ov = __shfl_xor(v, off, 32);
                int   oi = __shfl_xor(ii, off, 32);
                if (ov < v || (ov == v && oi < ii)) { v = ov; ii = oi; }
            }
            if (l15 == 0) {
                int rowL = 32 * wm + 16 * rf + j + hl * 8;  // lanes 0/16 hold M=j / M=j+8
                redv[rowL][wc] = v;
                redi[rowL][wc] = ii;
            }
        }
    __syncthreads();
    if (t < MT) {
        float v0 = redv[t][0], v1 = redv[t][1];
        int   i0 = redi[t][0], i1 = redi[t][1];
        fidx[t] = (v1 < v0 || (v1 == v0 && i1 < i0)) ? i1 : i0;
    }
    __syncthreads();

    // fused gather + straight-through output + loss (pure f32, matches reference)
    float ls = 0.f;
#pragma unroll 4
    for (int i = 0; i < (MT * (D / 4)) / 256; ++i) {   // 64 float4 iterations
        int v   = t + i * 256;
        int row = v >> 7;                               // 128 float4 per row
        int d4  = v & 127;
        int code = fidx[row];
        float4 wv = ((const float4*)(weight + (size_t)code * D))[d4];
        float4 xv = ((const float4*)(x + (size_t)(row0g + row) * D))[d4];
        float dx = wv.x - xv.x, dy = wv.y - xv.y, dz = wv.z - xv.z, dw = wv.w - xv.w;
        ls += dx * dx + dy * dy + dz * dz + dw * dw;
        float4 o = { xv.x + dx, xv.y + dy, xv.z + dz, xv.w + dw };  // x + (quant-x)
        ((float4*)(out + (size_t)(row0g + row) * D))[d4] = o;
    }
#pragma unroll
    for (int off = 16; off; off >>= 1) ls += __shfl_xor(ls, off, 32);
    if (lane == 0) lred[wave] = ls;
    __syncthreads();
    if (t == 0) {
        float s = 0.f;
#pragma unroll
        for (int i = 0; i < 8; ++i) s += lred[i];
        // loss = (1 + 0.25) * mean((quant - x)^2)
        atomicAdd(loss, s * (1.25f / 16777216.f));
    }
}

// ---- Host launcher ----------------------------------------------------------
extern "C" void kernel_launch(void* const* d_in, const int* in_sizes, int n_in,
                              void* d_out, int out_size, void* d_ws, size_t ws_size,
                              hipStream_t stream) {
    const float* x = (const float*)d_in[0];      // 32*1024*512 f32
    const float* w = (const float*)d_in[1];      // 8192*512 f32
    float* out  = (float*)d_out;                 // quant_st (16777216) + loss (1)
    float* loss = out + (size_t)NROW * D;

    uint8_t* ws = (uint8_t*)d_ws;                // workspace layout (80MB + 32KB)
    uint16_t* xhi = (uint16_t*)(ws);                        // 32 MB
    uint16_t* xlo = (uint16_t*)(ws + (size_t)33554432);     // 32 MB
    uint16_t* whi = (uint16_t*)(ws + (size_t)67108864);     //  8 MB
    uint16_t* wlo = (uint16_t*)(ws + (size_t)75497472);     //  8 MB
    float*  wnorm = (float*)  (ws + (size_t)83886080);      // 32 KB

    vq_cvt_x<<<(NROW * D) / (256 * 4), 256, 0, stream>>>(x, xhi, xlo);
    vq_cvt_w<<<K / 8, 256, 0, stream>>>(w, whi, wlo, wnorm);
    vq_zero<<<1, 32, 0, stream>>>(loss);
    vq_main<<<NROW / MT, 256, 0, stream>>>(x, w, xhi, xlo, whi, wlo, wnorm, out, loss);
}
